// BEVFormer_69209103008009
// MI455X (gfx1250) — matrix operand order repaired and checked by reading, not compiled, not run on hardware
//
#include <hip/hip_runtime.h>
#include <hip/hip_bf16.h>
#include <math.h>

typedef __attribute__((ext_vector_type(16))) _Float16 v16h;
typedef __attribute__((ext_vector_type(8)))  float    v8f;

#define NCAM   6
#define Dm     256
#define NQ     2500
#define NHEADS 8
#define HD     32
#define NPTS   4
#define FH     28
#define FW     50
#define IMG_Hc 448
#define IMG_Wc 800
#define NSAMP  192      /* NCAM*NHEADS*NPTS */
#define EPSV   1e-5f

/* async global->LDS copy (CDNA5): per-lane LDS byte offset + 64b global addr */
__device__ __forceinline__ void async_g2l_b128(unsigned lds_off, const float* g)
{
    asm volatile("global_load_async_to_lds_b128 %0, %1, off"
                 :: "v"(lds_off), "v"(g) : "memory");
}
__device__ __forceinline__ void wait_asynccnt0()
{
    asm volatile("s_wait_asynccnt 0" ::: "memory");
}

/* ------------------------------------------------------------------ */
/* Tiled GEMM:  C[MxN] = act( alpha * A*B + bias + addsrc )            */
/* A: MxK row-major, B: KxN row-major. K must be a multiple of 32.     */
/* f16 inputs into v_wmma_f32_16x16x32_f16, f32 accumulate.            */
/* bias_mode: 0 none, 1 per-col(n), 2 per-row(m). act: 0/1 gelu/2 relu */
/* ------------------------------------------------------------------ */
__global__ __launch_bounds__(128)
void gemm_wmma_kernel(const float* __restrict__ A, const float* __restrict__ B,
                      const float* __restrict__ bias, const float* __restrict__ addsrc,
                      float* __restrict__ C,
                      int M, int N, int K, int lda, int ldb, int ldc,
                      int bias_mode, int act, float alpha)
{
    __shared__ _Float16 As[64][40];
    __shared__ _Float16 Bs[64][40];   /* stored transposed: Bs[n][k] */

    const int tid  = threadIdx.x;
    const int m0   = blockIdx.y * 64;
    const int n0   = blockIdx.x * 64;
    const int lane = tid & 31;
    const int wave = tid >> 5;
    const int wm   = (wave & 1) * 32;
    const int wn   = (wave >> 1) * 32;
    const bool interior = (m0 + 64 <= M) && (n0 + 64 <= N);

    v8f c[2][2] = {};

    for (int k0 = 0; k0 < K; k0 += 32) {
        if (interior) {
            /* fully in-bounds tile: vectorized 16B loads, f32->f16, LDS store */
#pragma unroll
            for (int it = 0; it < 4; ++it) {
                const int e = it * 128 + tid;            /* 0..511 */
                /* A tile 64x32: 8 float4 chunks per row, lanes sweep K first */
                const int am = e >> 3, ak = (e & 7) * 4;
                const float4 av = *(const float4*)(A + (size_t)(m0 + am) * lda + (k0 + ak));
                As[am][ak + 0] = (_Float16)av.x;
                As[am][ak + 1] = (_Float16)av.y;
                As[am][ak + 2] = (_Float16)av.z;
                As[am][ak + 3] = (_Float16)av.w;
                /* B tile 32x64: 16 float4 chunks per k-row, store transposed */
                const int bk = e >> 4, bn = (e & 15) * 4;
                const float4 bv = *(const float4*)(B + (size_t)(k0 + bk) * ldb + (n0 + bn));
                Bs[bn + 0][bk] = (_Float16)bv.x;
                Bs[bn + 1][bk] = (_Float16)bv.y;
                Bs[bn + 2][bk] = (_Float16)bv.z;
                Bs[bn + 3][bk] = (_Float16)bv.w;
            }
        } else {
            /* edge tile: guarded scalar path (K tiles are always full) */
            for (int idx = tid; idx < 2048; idx += 128) {
                const int mm = idx >> 5;
                const int kk = idx & 31;
                const int gm = m0 + mm, gn = n0 + mm, gk = k0 + kk;
                As[mm][kk] = (_Float16)((gm < M) ? A[(size_t)gm * lda + gk] : 0.f);
                Bs[mm][kk] = (_Float16)((gn < N) ? B[(size_t)gk * ldb + gn] : 0.f);
            }
        }
        __syncthreads();

        const int r  = lane & 15;
        const int hi = lane >> 4;
        v16h af[2], bf[2];
        /* A frag: element h -> K = 16*(h>>3) + 8*hi + (h&7), M = r (ISA 7.12.2) */
#pragma unroll
        for (int mi = 0; mi < 2; ++mi)
#pragma unroll
            for (int h = 0; h < 16; ++h)
                af[mi][h] = As[wm + mi * 16 + r][((h >> 3) << 4) + (hi << 3) + (h & 7)];
        /* B frag: element h -> K = 16*(h>>3) + 8*hi + (h&7), N = r */
#pragma unroll
        for (int ni = 0; ni < 2; ++ni)
#pragma unroll
            for (int h = 0; h < 16; ++h)
                bf[ni][h] = Bs[wn + ni * 16 + r][((h >> 3) << 4) + (hi << 3) + (h & 7)];

#pragma unroll
        for (int mi = 0; mi < 2; ++mi)
#pragma unroll
            for (int ni = 0; ni < 2; ++ni)
                c[mi][ni] = __builtin_amdgcn_wmma_f32_16x16x32_f16(
                    false, af[mi], false, bf[ni], (short)0, c[mi][ni], false, false);
        __syncthreads();
    }

    /* epilogue: C layout VGPR rr -> M = rr + 8*hi, N = lane&15 */
    const int r  = lane & 15;
    const int hi = lane >> 4;
#pragma unroll
    for (int mi = 0; mi < 2; ++mi)
#pragma unroll
        for (int ni = 0; ni < 2; ++ni)
#pragma unroll
            for (int rr = 0; rr < 8; ++rr) {
                const int m = m0 + wm + mi * 16 + rr + hi * 8;
                const int n = n0 + wn + ni * 16 + r;
                if (m < M && n < N) {
                    float v = c[mi][ni][rr] * alpha;
                    if (bias_mode == 1) v += bias[n];
                    else if (bias_mode == 2) v += bias[m];
                    if (addsrc) v += addsrc[(size_t)m * ldc + n];
                    if (act == 1) v = 0.5f * v * (1.f + erff(v * 0.70710678118654752f));
                    else if (act == 2) v = fmaxf(v, 0.f);
                    C[(size_t)m * ldc + n] = v;
                }
            }
}

/* ------------------------------------------------------------------ */
/* Direct 3x3 stride-2 pad-1 convolution (single image, NCHW)          */
/* ------------------------------------------------------------------ */
__global__ __launch_bounds__(256)
void conv3x3s2_kernel(const float* __restrict__ in, const float* __restrict__ w,
                      const float* __restrict__ bias, float* __restrict__ out,
                      int Ci, int Hi, int Wi, int Co, int Ho, int Wo)
{
    const size_t idx   = (size_t)blockIdx.x * 256 + threadIdx.x;
    const size_t total = (size_t)Co * Ho * Wo;
    if (idx >= total) return;
    const int ox = (int)(idx % Wo);
    const int t1 = (int)(idx / Wo);
    const int oy = t1 % Ho;
    const int co = t1 / Ho;

    float acc = bias[co];
    const int iy0 = oy * 2 - 1;
    const int ix0 = ox * 2 - 1;
    const float* wp = w + (size_t)co * Ci * 9;
    for (int ci = 0; ci < Ci; ++ci) {
        const float* ip = in + (size_t)ci * Hi * Wi;
        const float* wc = wp + ci * 9;
        if (ci + 1 < Ci) {  /* keep next input plane warm (global_prefetch_b8) */
            const int py = (iy0 < 0) ? 0 : iy0;
            const int px = (ix0 < 0) ? 0 : ix0;
            __builtin_prefetch(ip + (size_t)Hi * Wi + (size_t)py * Wi + px, 0, 1);
        }
#pragma unroll
        for (int ky = 0; ky < 3; ++ky) {
            const int iy = iy0 + ky;
            if (iy < 0 || iy >= Hi) continue;
            const float* row = ip + (size_t)iy * Wi;
#pragma unroll
            for (int kx = 0; kx < 3; ++kx) {
                const int ix = ix0 + kx;
                if (ix < 0 || ix >= Wi) continue;
                acc += row[ix] * wc[ky * 3 + kx];
            }
        }
    }
    out[idx] = acc;
}

/* ------------------------------------------------------------------ */
/* GroupNorm: stats (one block per group) + apply (elementwise)        */
/* ------------------------------------------------------------------ */
__global__ __launch_bounds__(256)
void gn_stats_kernel(const float* __restrict__ x, float* __restrict__ stat,
                     int chpg, int HW)
{
    __shared__ float ssum[256];
    __shared__ float ssq[256];
    const int g = blockIdx.x;
    const size_t n = (size_t)chpg * HW;
    const float* xp = x + (size_t)g * n;
    float s = 0.f, s2 = 0.f;
    for (size_t i = threadIdx.x; i < n; i += 256) {
        const float v = xp[i];
        s += v; s2 += v * v;
    }
    ssum[threadIdx.x] = s; ssq[threadIdx.x] = s2;
    __syncthreads();
    for (int o = 128; o > 0; o >>= 1) {
        if ((int)threadIdx.x < o) {
            ssum[threadIdx.x] += ssum[threadIdx.x + o];
            ssq[threadIdx.x]  += ssq[threadIdx.x + o];
        }
        __syncthreads();
    }
    if (threadIdx.x == 0) {
        const float mean = ssum[0] / (float)n;
        const float var  = ssq[0] / (float)n - mean * mean;
        stat[g * 2 + 0] = mean;
        stat[g * 2 + 1] = fmaxf(var, 0.f);
    }
}

__global__ __launch_bounds__(256)
void gn_apply_kernel(float* __restrict__ x, const float* __restrict__ stat,
                     const float* __restrict__ gamma, const float* __restrict__ beta,
                     int C, int HW, int chpg, int relu)
{
    const size_t idx = (size_t)blockIdx.x * 256 + threadIdx.x;
    if (idx >= (size_t)C * HW) return;
    const int c = (int)(idx / HW);
    const int g = c / chpg;
    const float mean = stat[g * 2 + 0];
    const float var  = stat[g * 2 + 1];
    float v = (x[idx] - mean) * rsqrtf(var + EPSV) * gamma[c] + beta[c];
    if (relu) v = fmaxf(v, 0.f);
    x[idx] = v;
}

/* ------------------------------------------------------------------ */
/* 32x32 LDS-tiled transpose: out[c][r] = in[r][c], in is R x Ccols    */
/* ------------------------------------------------------------------ */
__global__ __launch_bounds__(256)
void transpose_kernel(const float* __restrict__ in, float* __restrict__ out,
                      int R, int Ccols)
{
    __shared__ float tile[32][33];
    const int bx = blockIdx.x * 32;   /* column block */
    const int by = blockIdx.y * 32;   /* row block */
    const int tx = threadIdx.x & 31;
    const int ty = (int)threadIdx.x >> 5;  /* 8 rows per pass */
#pragma unroll
    for (int j = 0; j < 32; j += 8) {
        const int r = by + ty + j, cc = bx + tx;
        tile[ty + j][tx] = (r < R && cc < Ccols) ? in[(size_t)r * Ccols + cc] : 0.f;
    }
    __syncthreads();
#pragma unroll
    for (int j = 0; j < 32; j += 8) {
        const int cc = bx + ty + j, r = by + tx;
        if (cc < Ccols && r < R) out[(size_t)cc * R + r] = tile[tx][ty + j];
    }
}

/* ------------------------------------------------------------------ */
/* LayerNorm over last dim (256), one wave32 per row                   */
/* ------------------------------------------------------------------ */
__global__ __launch_bounds__(256)
void layernorm_kernel(const float* __restrict__ x, const float* __restrict__ g,
                      const float* __restrict__ b, float* __restrict__ y, int nrows)
{
    const int row  = blockIdx.x * 8 + ((int)threadIdx.x >> 5);
    const int lane = threadIdx.x & 31;
    if (row >= nrows) return;
    const float* xr = x + (size_t)row * Dm;
    float v[8];
    float s = 0.f;
#pragma unroll
    for (int j = 0; j < 8; ++j) { v[j] = xr[lane + j * 32]; s += v[j]; }
    for (int o = 16; o > 0; o >>= 1) s += __shfl_xor(s, o, 32);
    const float mean = s * (1.f / Dm);
    float ss = 0.f;
#pragma unroll
    for (int j = 0; j < 8; ++j) { const float d = v[j] - mean; ss += d * d; }
    for (int o = 16; o > 0; o >>= 1) ss += __shfl_xor(ss, o, 32);
    const float inv = rsqrtf(ss * (1.f / Dm) + EPSV);
    float* yr = y + (size_t)row * Dm;
#pragma unroll
    for (int j = 0; j < 8; ++j) {
        const int col = lane + j * 32;
        yr[col] = (v[j] - mean) * inv * g[col] + b[col];
    }
}

/* ------------------------------------------------------------------ */
/* Fused MHA (flash-style online softmax).                             */
/* K/V tiles staged via GLOBAL_LOAD_ASYNC_TO_LDS_B128 (ASYNCcnt).      */
/* ------------------------------------------------------------------ */
__global__ __launch_bounds__(256)
void mha_kernel(const float* __restrict__ Q, const float* __restrict__ Kx,
                const float* __restrict__ Vx, float* __restrict__ O)
{
    __shared__ float Ks[64][HD];
    __shared__ float Vs[64][HD];
    const int h    = blockIdx.y;
    const int qi   = blockIdx.x * 256 + threadIdx.x;
    const int col0 = h * HD;
    float qreg[HD];
    if (qi < NQ) {
#pragma unroll
        for (int j = 0; j < HD; ++j) qreg[j] = Q[(size_t)qi * Dm + col0 + j];
    }
    float m = -INFINITY, l = 0.f;
    float acc[HD] = {};
    const float scale = 0.17677669529663687f; /* 1/sqrt(32) */

    for (int kt = 0; kt < NQ; kt += 64) {
        const int kmax = min(64, NQ - kt);
        __syncthreads();
        /* 2 x b128 per thread per array; rows >= kmax may stage garbage but
           are never consumed (reads stay inside the workspace allocation). */
#pragma unroll
        for (int p = 0; p < 2; ++p) {
            const int e  = p * 1024 + (int)threadIdx.x * 4;
            const int rr = e >> 5;
            const int jj = e & 31;
            async_g2l_b128((unsigned)(size_t)&Ks[rr][jj],
                           Kx + (size_t)(kt + rr) * Dm + col0 + jj);
            async_g2l_b128((unsigned)(size_t)&Vs[rr][jj],
                           Vx + (size_t)(kt + rr) * Dm + col0 + jj);
        }
        wait_asynccnt0();
        __syncthreads();
        if (qi < NQ) {
            for (int kk = 0; kk < kmax; ++kk) {
                float s = 0.f;
#pragma unroll
                for (int j = 0; j < HD; ++j) s += qreg[j] * Ks[kk][j];
                s *= scale;
                const float nm   = fmaxf(m, s);
                const float corr = expf(m - nm);
                const float p    = expf(s - nm);
                l = l * corr + p;
#pragma unroll
                for (int j = 0; j < HD; ++j) acc[j] = acc[j] * corr + p * Vs[kk][j];
                m = nm;
            }
        }
    }
    if (qi < NQ) {
        const float inv = 1.f / l;
#pragma unroll
        for (int j = 0; j < HD; ++j) O[(size_t)qi * Dm + col0 + j] = acc[j] * inv;
    }
}

/* ------------------------------------------------------------------ */
/* Camera projection of BEV grid points                                */
/* ------------------------------------------------------------------ */
__global__ __launch_bounds__(256)
void project_kernel(const float* __restrict__ Kmat, const float* __restrict__ Emat,
                    float* __restrict__ ref, int* __restrict__ valid)
{
    const int q = blockIdx.x * 256 + threadIdx.x;
    if (q >= NQ) return;
    const int i = q / 50, j = q % 50;
    const float px = ((float)i - 24.5f) * 0.5f;
    const float py = ((float)j - 24.5f) * 0.5f;
    for (int c = 0; c < NCAM; ++c) {
        const float* E  = Emat + c * 16;
        const float* Kc = Kmat + c * 9;
        const float pc0 = E[0] * px + E[1] * py + E[3];
        const float pc1 = E[4] * px + E[5] * py + E[7];
        const float pc2 = E[8] * px + E[9] * py + E[11];
        const float x0 = Kc[0] * pc0 + Kc[1] * pc1 + Kc[2] * pc2;
        const float x1 = Kc[3] * pc0 + Kc[4] * pc1 + Kc[5] * pc2;
        const float x2 = Kc[6] * pc0 + Kc[7] * pc1 + Kc[8] * pc2;
        const float z = fmaxf(x2, 1e-5f);
        const float u = x0 / z, v = x1 / z;
        const int ok = (pc2 > 0.f) && (u >= 0.f) && (u < (float)IMG_Wc) &&
                       (v >= 0.f) && (v < (float)IMG_Hc);
        ref[((size_t)c * NQ + q) * 2 + 0] = 2.f * u / (IMG_Wc - 1) - 1.f;
        ref[((size_t)c * NQ + q) * 2 + 1] = 2.f * v / (IMG_Hc - 1) - 1.f;
        valid[(size_t)c * NQ + q] = ok;
    }
}

/* ------------------------------------------------------------------ */
/* Masked softmax over 192 cross-attn logits, one wave32 per query     */
/* ------------------------------------------------------------------ */
__global__ __launch_bounds__(256)
void sca_softmax_kernel(float* __restrict__ logits, const int* __restrict__ valid)
{
    const int row  = blockIdx.x * 8 + ((int)threadIdx.x >> 5);
    const int lane = threadIdx.x & 31;
    if (row >= NQ) return;
    float* lr = logits + (size_t)row * NSAMP;
    float lv[NCAM];
    int   ok[NCAM];
    float mx = -INFINITY;
#pragma unroll
    for (int c = 0; c < NCAM; ++c) {
        ok[c] = valid[(size_t)c * NQ + row];
        const float x = lr[c * 32 + lane];
        lv[c] = ok[c] ? x : -1e30f;
        mx = fmaxf(mx, lv[c]);
    }
    for (int o = 16; o > 0; o >>= 1) mx = fmaxf(mx, __shfl_xor(mx, o, 32));
    float s = 0.f;
#pragma unroll
    for (int c = 0; c < NCAM; ++c) { lv[c] = expf(lv[c] - mx); s += lv[c]; }
    for (int o = 16; o > 0; o >>= 1) s += __shfl_xor(s, o, 32);
    const float inv = 1.f / s;
#pragma unroll
    for (int c = 0; c < NCAM; ++c)
        lr[c * 32 + lane] = ok[c] ? lv[c] * inv : 0.f;
}

/* ------------------------------------------------------------------ */
/* Deformable sampling: bilinear gather + weighted sum.                */
/* grid = NQ blocks, 256 threads (one per output channel).             */
/* vals layout: [cam][hw][256] (channel-minor, coalesced gathers)      */
/* ------------------------------------------------------------------ */
__global__ __launch_bounds__(256)
void sca_sample_kernel(const float* __restrict__ offs, const float* __restrict__ wts,
                       const float* __restrict__ ref, const float* __restrict__ vals,
                       float* __restrict__ scat)
{
    __shared__ float so[NSAMP * 2];
    __shared__ float sw[NSAMP];
    __shared__ float sref[NCAM * 2];
    const int q = blockIdx.x;
    const int t = threadIdx.x;
    for (int i = t; i < NSAMP * 2; i += 256) so[i] = offs[(size_t)q * NSAMP * 2 + i];
    for (int i = t; i < NSAMP; i += 256)     sw[i] = wts[(size_t)q * NSAMP + i];
    if (t < NCAM * 2) sref[t] = ref[((size_t)(t >> 1) * NQ + q) * 2 + (t & 1)];
    __syncthreads();

    float acc = 0.f;
    const float sclx = 2.f / (FW - 1);
    const float scly = 2.f / (FH - 1);
    for (int s = 0; s < NSAMP; ++s) {
        const float w = sw[s];
        if (w == 0.f) continue;           /* uniform across block */
        const int c = s >> 5;
        const float px = sref[c * 2 + 0] + so[s * 2 + 0] * sclx;
        const float py = sref[c * 2 + 1] + so[s * 2 + 1] * scly;
        const float gx = (px + 1.f) * 0.5f * (FW - 1);
        const float gy = (py + 1.f) * 0.5f * (FH - 1);
        const float x0f = floorf(gx), y0f = floorf(gy);
        const float wx = gx - x0f, wy = gy - y0f;
        const int x0 = (int)x0f, y0 = (int)y0f;
        const float* vb = vals + (size_t)c * FH * FW * Dm;
        float sampled = 0.f;
#pragma unroll
        for (int corner = 0; corner < 4; ++corner) {
            const int dx = corner & 1, dy = corner >> 1;
            const int xi = x0 + dx, yi = y0 + dy;
            if (xi >= 0 && xi < FW && yi >= 0 && yi < FH) {
                const float cw = (dx ? wx : 1.f - wx) * (dy ? wy : 1.f - wy);
                sampled += cw * vb[((size_t)yi * FW + xi) * Dm + t];
            }
        }
        acc += w * sampled;
    }
    scat[(size_t)q * Dm + t] = acc;
}

/* ------------------------------------------------------------------ */
__global__ __launch_bounds__(256)
void add2_kernel(const float* __restrict__ a, const float* __restrict__ b,
                 float* __restrict__ y, int n)
{
    const int i = blockIdx.x * 256 + threadIdx.x;
    if (i < n) y[i] = a[i] + b[i];
}

__global__ __launch_bounds__(256)
void copy_kernel(const float* __restrict__ a, float* __restrict__ y, int n)
{
    const int i = blockIdx.x * 256 + threadIdx.x;
    if (i < n) y[i] = a[i];
}

/* ================================================================== */
extern "C" void kernel_launch(void* const* d_in, const int* in_sizes, int n_in,
                              void* d_out, int out_size, void* d_ws, size_t ws_size,
                              hipStream_t stream)
{
    (void)in_sizes; (void)n_in; (void)out_size; (void)ws_size;

    const float* images  = (const float*)d_in[0];
    const float* intr    = (const float*)d_in[1];
    const float* ego     = (const float*)d_in[2];
    const float* bev_q   = (const float*)d_in[3];
    const float* bev_pos = (const float*)d_in[4];
    const float* bb_w[5]    = {(const float*)d_in[5],  (const float*)d_in[9],
                               (const float*)d_in[13], (const float*)d_in[17],
                               (const float*)d_in[21]};
    const float* bb_b[5]    = {(const float*)d_in[6],  (const float*)d_in[10],
                               (const float*)d_in[14], (const float*)d_in[18],
                               (const float*)d_in[22]};
    const float* bb_g[5]    = {(const float*)d_in[7],  (const float*)d_in[11],
                               (const float*)d_in[15], (const float*)d_in[19],
                               (const float*)d_in[23]};
    const float* bb_beta[5] = {(const float*)d_in[8],  (const float*)d_in[12],
                               (const float*)d_in[16], (const float*)d_in[20],
                               (const float*)d_in[24]};
    const float* sa_wq = (const float*)d_in[25]; const float* sa_bq = (const float*)d_in[26];
    const float* sa_wk = (const float*)d_in[27]; const float* sa_bk = (const float*)d_in[28];
    const float* sa_wv = (const float*)d_in[29]; const float* sa_bv = (const float*)d_in[30];
    const float* sa_wo = (const float*)d_in[31]; const float* sa_bo = (const float*)d_in[32];
    const float* ln1_g = (const float*)d_in[33]; const float* ln1_b = (const float*)d_in[34];
    const float* ln2_g = (const float*)d_in[35]; const float* ln2_b = (const float*)d_in[36];
    const float* ln3_g = (const float*)d_in[37]; const float* ln3_b = (const float*)d_in[38];
    const float* off_w = (const float*)d_in[39]; const float* off_b = (const float*)d_in[40];
    const float* wt_w  = (const float*)d_in[41]; const float* wt_b  = (const float*)d_in[42];
    const float* val_w = (const float*)d_in[43]; const float* val_b = (const float*)d_in[44];
    const float* out_w = (const float*)d_in[45]; const float* out_b = (const float*)d_in[46];
    const float* ffn_w1 = (const float*)d_in[47]; const float* ffn_b1 = (const float*)d_in[48];
    const float* ffn_w2 = (const float*)d_in[49]; const float* ffn_b2 = (const float*)d_in[50];

    /* ---- workspace carving ---- */
    char* ws = (char*)d_ws;
    size_t off = 0;
    auto carve = [&](size_t nbytes) -> char* {
        char* p = ws + off;
        off = (off + nbytes + 255) & ~(size_t)255;
        return p;
    };
    float* bufA   = (float*)carve((size_t)64 * 224 * 400 * 4);    /* max conv0/conv2 out */
    float* bufB   = (float*)carve((size_t)128 * 112 * 200 * 4);   /* max conv1/conv3 out */
    float* feats  = (float*)carve((size_t)NCAM * Dm * FH * FW * 4);
    float* featsT = (float*)carve((size_t)NCAM * FH * FW * Dm * 4);
    float* vals   = (float*)carve((size_t)NCAM * FH * FW * Dm * 4);
    float* gnstat = (float*)carve(64 * 4);
    float* refb   = (float*)carve((size_t)NCAM * NQ * 2 * 4);
    int*   validb = (int*)  carve((size_t)NCAM * NQ * 4);
    float* qb     = (float*)carve((size_t)NQ * Dm * 4);
    float* xq     = (float*)carve((size_t)NQ * Dm * 4);
    float* tq     = (float*)carve((size_t)NQ * Dm * 4);
    float* tk     = (float*)carve((size_t)NQ * Dm * 4);
    float* tv     = (float*)carve((size_t)NQ * Dm * 4);
    float* attnb  = (float*)carve((size_t)NQ * Dm * 4);
    float* scat   = (float*)carve((size_t)NQ * Dm * 4);
    float* offsb  = (float*)carve((size_t)NQ * NSAMP * 2 * 4);
    float* wtsb   = (float*)carve((size_t)NQ * NSAMP * 4);
    float* ffnh   = (float*)carve((size_t)NQ * 1024 * 4);

    auto gemm = [&](const float* A, const float* B, const float* bias,
                    const float* addsrc, float* C, int M, int N, int K,
                    int lda, int ldb, int ldc, int bias_mode, int act, float alpha) {
        dim3 g((N + 63) / 64, (M + 63) / 64);
        gemm_wmma_kernel<<<g, dim3(128), 0, stream>>>(
            A, B, bias, addsrc, C, M, N, K, lda, ldb, ldc, bias_mode, act, alpha);
    };
    auto groupnorm = [&](float* x, int C, int HW, const float* g, const float* bt, int relu) {
        gn_stats_kernel<<<32, 256, 0, stream>>>(x, gnstat, C / 32, HW);
        const size_t tot = (size_t)C * HW;
        gn_apply_kernel<<<(unsigned)((tot + 255) / 256), 256, 0, stream>>>(
            x, gnstat, g, bt, C, HW, C / 32, relu);
    };

    /* ---- backbone, per camera ---- */
    struct CL { int Ci, Hi, Wi, Co, Ho, Wo; };
    const CL L[4] = {{3, 448, 800, 64, 224, 400},
                     {64, 224, 400, 128, 112, 200},
                     {128, 112, 200, 256, 56, 100},
                     {256, 56, 100, 256, 28, 50}};
    for (int cam = 0; cam < NCAM; ++cam) {
        const float* src = images + (size_t)cam * 3 * IMG_Hc * IMG_Wc;
        const float* inp[4]  = {src, bufA, bufB, bufA};
        float*       outp[4] = {bufA, bufB, bufA, bufB};
        for (int li = 0; li < 4; ++li) {
            const CL& c = L[li];
            const size_t tot = (size_t)c.Co * c.Ho * c.Wo;
            conv3x3s2_kernel<<<(unsigned)((tot + 255) / 256), 256, 0, stream>>>(
                inp[li], bb_w[li], bb_b[li], outp[li],
                c.Ci, c.Hi, c.Wi, c.Co, c.Ho, c.Wo);
            groupnorm(outp[li], c.Co, c.Ho * c.Wo, bb_g[li], bb_beta[li], 1);
        }
        /* conv4 (1x1) as WMMA GEMM: feats[cam] = w4(256x256) @ act3(256x1400) + b4 */
        float* featc = feats + (size_t)cam * Dm * FH * FW;
        gemm(bb_w[4], bufB, bb_b[4], nullptr, featc,
             Dm, FH * FW, Dm, Dm, FH * FW, FH * FW, /*bias per-m*/2, 0, 1.f);
        groupnorm(featc, Dm, FH * FW, bb_g[4], bb_beta[4], 0);
        /* featsT[cam][hw][c] for the value-projection GEMM (row-major A) */
        transpose_kernel<<<dim3((FH * FW + 31) / 32, (Dm + 31) / 32), 256, 0, stream>>>(
            featc, featsT + (size_t)cam * FH * FW * Dm, Dm, FH * FW);
    }

    /* ---- geometry + query init ---- */
    project_kernel<<<(NQ + 255) / 256, 256, 0, stream>>>(intr, ego, refb, validb);
    add2_kernel<<<(NQ * Dm + 255) / 256, 256, 0, stream>>>(bev_q, bev_pos, qb, NQ * Dm);

    /* ---- transformer layers ---- */
    for (int l = 0; l < 2; ++l) {
        const float* wq = sa_wq + (size_t)l * Dm * Dm; const float* bq = sa_bq + (size_t)l * Dm;
        const float* wk = sa_wk + (size_t)l * Dm * Dm; const float* bk = sa_bk + (size_t)l * Dm;
        const float* wv = sa_wv + (size_t)l * Dm * Dm; const float* bv = sa_bv + (size_t)l * Dm;
        const float* wo = sa_wo + (size_t)l * Dm * Dm; const float* bo = sa_bo + (size_t)l * Dm;
        const float* ow = off_w + (size_t)l * Dm * NSAMP * 2; const float* ob = off_b + (size_t)l * NSAMP * 2;
        const float* ww = wt_w + (size_t)l * Dm * NSAMP;      const float* wb = wt_b + (size_t)l * NSAMP;
        const float* vw = val_w + (size_t)l * Dm * Dm;        const float* vb = val_b + (size_t)l * Dm;
        const float* pw = out_w + (size_t)l * Dm * Dm;        const float* pb = out_b + (size_t)l * Dm;
        const float* f1 = ffn_w1 + (size_t)l * Dm * 1024;     const float* fb1 = ffn_b1 + (size_t)l * 1024;
        const float* f2 = ffn_w2 + (size_t)l * 1024 * Dm;     const float* fb2 = ffn_b2 + (size_t)l * Dm;

        /* 1) self-attention */
        layernorm_kernel<<<(NQ + 7) / 8, 256, 0, stream>>>(qb, ln1_g + l * Dm, ln1_b + l * Dm, xq, NQ);
        gemm(xq, wq, bq, nullptr, tq, NQ, Dm, Dm, Dm, Dm, Dm, 1, 0, 1.f);
        gemm(xq, wk, bk, nullptr, tk, NQ, Dm, Dm, Dm, Dm, Dm, 1, 0, 1.f);
        gemm(xq, wv, bv, nullptr, tv, NQ, Dm, Dm, Dm, Dm, Dm, 1, 0, 1.f);
        mha_kernel<<<dim3((NQ + 255) / 256, NHEADS), 256, 0, stream>>>(tq, tk, tv, attnb);
        gemm(attnb, wo, bo, qb, qb, NQ, Dm, Dm, Dm, Dm, Dm, 1, 0, 1.f);

        /* 2) spatial cross-attention */
        layernorm_kernel<<<(NQ + 7) / 8, 256, 0, stream>>>(qb, ln2_g + l * Dm, ln2_b + l * Dm, xq, NQ);
        gemm(xq, ow, ob, nullptr, offsb, NQ, NSAMP * 2, Dm, Dm, NSAMP * 2, NSAMP * 2, 1, 0, 1.f);
        gemm(xq, ww, wb, nullptr, wtsb, NQ, NSAMP, Dm, Dm, NSAMP, NSAMP, 1, 0, 1.f);
        sca_softmax_kernel<<<(NQ + 7) / 8, 256, 0, stream>>>(wtsb, validb);
        for (int cam = 0; cam < NCAM; ++cam) {
            /* vals[cam] = featsT[cam] (1400x256) @ vw (256x256) + vb */
            gemm(featsT + (size_t)cam * FH * FW * Dm, vw, vb, nullptr,
                 vals + (size_t)cam * FH * FW * Dm,
                 FH * FW, Dm, Dm, Dm, Dm, Dm, 1, 0, 1.f);
        }
        sca_sample_kernel<<<NQ, 256, 0, stream>>>(offsb, wtsb, refb, vals, scat);
        gemm(scat, pw, pb, qb, qb, NQ, Dm, Dm, Dm, Dm, Dm, 1, 0, 1.f);

        /* 3) FFN */
        layernorm_kernel<<<(NQ + 7) / 8, 256, 0, stream>>>(qb, ln3_g + l * Dm, ln3_b + l * Dm, xq, NQ);
        gemm(xq, f1, fb1, nullptr, ffnh, NQ, 1024, Dm, Dm, 1024, 1024, 1, /*gelu*/1, 1.f);
        gemm(ffnh, f2, fb2, qb, qb, NQ, Dm, 1024, 1024, Dm, Dm, 1, 0, 1.f);
    }

    copy_kernel<<<(NQ * Dm + 255) / 256, 256, 0, stream>>>(qb, (float*)d_out, NQ * Dm);
}
